// PGWCrossAttention_31817117728909
// MI455X (gfx1250) — compile-verified, matrix operand order checked by reading, not run on hardware
//
#include <hip/hip_runtime.h>
#include <hip/hip_bf16.h>
#include <math.h>

// Problem dims (match reference)
#define B_   8
#define NQ_  2048
#define NK_  2048
#define DIN_ 256
#define H_   32

#define LOG2E_ 1.4426950408889634f
#define LN2_   0.6931471805599453f

typedef _Float16 v16h __attribute__((ext_vector_type(16)));
typedef _Float16 v8h  __attribute__((ext_vector_type(8)));
typedef float    v8f  __attribute__((ext_vector_type(8)));

// ---- fast native transcendental helpers (single v_* trans ops on CDNA5) ----
static __device__ __forceinline__ float fexp2(float x) { return __builtin_amdgcn_exp2f(x); }
static __device__ __forceinline__ float flog2(float x) { return __builtin_amdgcn_logf(x); }
static __device__ __forceinline__ float fsqrt(float x) { return __builtin_amdgcn_sqrtf(x); }
static __device__ __forceinline__ float frcp(float x)  { return __builtin_amdgcn_rcpf(x); }

static __device__ __forceinline__ float ftanh(float x) {
  // tanh(x) = (e^{2x}-1)/(e^{2x}+1), via exp2; clamp to avoid inf*0
  float xc = fminf(fmaxf(x, -20.f), 20.f);
  float e = fexp2(2.0f * LOG2E_ * xc);
  return (e - 1.0f) * frcp(e + 1.0f);
}

static __device__ __forceinline__ v8f zero8() {
  v8f z = {0.f,0.f,0.f,0.f,0.f,0.f,0.f,0.f};
  return z;
}

static __device__ __forceinline__ v8f wmma16(v16h a, v16h b, v8f c) {
  // D = A(16x32 f16) * B(32x16 f16) + C(16x16 f32)
  return __builtin_amdgcn_wmma_f32_16x16x32_f16(false, a, false, b, (short)0, c,
                                                false, false);
}

// Load a 16-bit A/B fragment from a row-major tile (ld = 32 halves per row).
// Per CDNA5 layout: lane r=lane&15 supplies row r; half hf=lane>>4 selects
// K = 8*hf + {0..7} (VGPR 0-3) and K = 16 + 8*hf + {0..7} (VGPR 4-7).
static __device__ __forceinline__ v16h load_frag16(const _Float16* rowbase, int lane) {
  int hf = (lane >> 4) & 1;
  v8h lo = *(const v8h*)(rowbase + 8 * hf);
  v8h hi = *(const v8h*)(rowbase + 16 + 8 * hf);
  return __builtin_shufflevector(lo, hi, 0,1,2,3,4,5,6,7,8,9,10,11,12,13,14,15);
}

// Row reductions for C-fragments: each 16x16 row lives in one 16-lane half.
static __device__ __forceinline__ float redmax16(float v) {
  v = fmaxf(v, __shfl_xor(v, 1, 32));
  v = fmaxf(v, __shfl_xor(v, 2, 32));
  v = fmaxf(v, __shfl_xor(v, 4, 32));
  v = fmaxf(v, __shfl_xor(v, 8, 32));
  return v;
}
static __device__ __forceinline__ float redsum16(float v) {
  v += __shfl_xor(v, 1, 32); v += __shfl_xor(v, 2, 32);
  v += __shfl_xor(v, 4, 32); v += __shfl_xor(v, 8, 32);
  return v;
}
static __device__ __forceinline__ float redsum32(float v) {
  v += __shfl_xor(v, 1, 32); v += __shfl_xor(v, 2, 32);
  v += __shfl_xor(v, 4, 32); v += __shfl_xor(v, 8, 32);
  v += __shfl_xor(v, 16, 32);
  return v;
}

// poincare log-map bridge on a 32-wide row distributed across the wave
static __device__ __forceinline__ float log_map32(float y) {
  float safe = 0.85f * ftanh(y);
  float n2 = redsum32(safe * safe);
  float n  = fsqrt(n2);
  float ns = fmaxf(n, 1e-15f);
  float t  = fminf(ns, 1.0f - 1e-5f);
  // artanh(t) = 0.5*ln((1+t)/(1-t)) = 0.5*ln2*log2((1+t)/(1-t))
  float ath = (0.5f * LN2_) * flog2((1.0f + t) * frcp(1.0f - t));
  return ath * safe * frcp(ns);
}

static __device__ __forceinline__ float proj32(const float* sx, const float* W,
                                               const float* bias, int lane) {
  float acc = bias[lane];
  #pragma unroll 8
  for (int k = 0; k < DIN_; ++k)
    acc = fmaf(sx[k], W[k * 32 + lane], acc);
  return acc;
}

static __device__ __forceinline__ float proj3(const float* sx, const float* W,
                                              const float* bias, int lane) {
  if (lane >= 3) return 0.f;
  float acc = bias[lane];
  #pragma unroll 8
  for (int k = 0; k < DIN_; ++k)
    acc = fmaf(sx[k], W[k * 3 + lane], acc);
  return acc;
}

// ---------------------------------------------------------------------------
// Kernel 1: query-side projections
// ---------------------------------------------------------------------------
__global__ void proj_q_kernel(const float* __restrict__ q_fp,
                              const float* __restrict__ Wqs, const float* __restrict__ bqs,
                              const float* __restrict__ Wqe, const float* __restrict__ bqe,
                              const float* __restrict__ Wqo, const float* __restrict__ bqo,
                              _Float16* __restrict__ Qs16, _Float16* __restrict__ Qe16,
                              float* __restrict__ Qo, float* __restrict__ qq) {
  __shared__ float sx[4][DIN_];
  int wave = threadIdx.x >> 5, lane = threadIdx.x & 31;
  long long row = (long long)blockIdx.x * 4 + wave;    // 0 .. B*NQ-1
  const float* x = q_fp + row * DIN_;
  for (int k = lane; k < DIN_; k += 32) sx[wave][k] = x[k];
  __syncthreads();
  float ys = proj32(sx[wave], Wqs, bqs, lane);
  float ye = proj32(sx[wave], Wqe, bqe, lane);
  float yo = proj3 (sx[wave], Wqo, bqo, lane);
  float qs = log_map32(ys);
  float qe = log_map32(ye);
  _Float16 qeh = (_Float16)qe;
  Qs16[row * 32 + lane] = (_Float16)qs;
  Qe16[row * 32 + lane] = qeh;
  float qef = (float)qeh;                       // match f16-rounded values in d2
  float s2 = redsum32(qef * qef);
  if (lane == 0) qq[row] = s2;
  float y0 = __shfl(yo, 0, 32), y1 = __shfl(yo, 1, 32), y2 = __shfl(yo, 2, 32);
  float s = y0 * y0 + y1 * y1 + y2 * y2;
  float n = fmaxf(fsqrt(s), 1e-12f);
  float invn = frcp(n);
  if (lane == 0) Qo[row * 4 + 0] = y0 * invn;
  if (lane == 1) Qo[row * 4 + 1] = y1 * invn;
  if (lane == 2) Qo[row * 4 + 2] = y2 * invn;
  if (lane == 3) Qo[row * 4 + 3] = s * invn * invn;  // ||normalized||^2
}

// ---------------------------------------------------------------------------
// Kernel 2: key/value-side projections (V stored transposed for P*V WMMA)
// ---------------------------------------------------------------------------
__global__ void proj_kv_kernel(const float* __restrict__ v_ret,
                               const float* __restrict__ Wks, const float* __restrict__ bks,
                               const float* __restrict__ Wvs, const float* __restrict__ bvs,
                               const float* __restrict__ Wke, const float* __restrict__ bke,
                               const float* __restrict__ Wve, const float* __restrict__ bve,
                               const float* __restrict__ Wko, const float* __restrict__ bko,
                               _Float16* __restrict__ Ks16, _Float16* __restrict__ Ke16,
                               _Float16* __restrict__ VsT16, _Float16* __restrict__ VeT16,
                               float* __restrict__ Ko, float* __restrict__ kk) {
  __shared__ float sx[4][DIN_];
  int wave = threadIdx.x >> 5, lane = threadIdx.x & 31;
  long long row = (long long)blockIdx.x * 4 + wave;    // 0 .. B*NK-1
  long long b   = row / NK_;
  long long key = row % NK_;
  const float* x = v_ret + row * DIN_;
  for (int k = lane; k < DIN_; k += 32) sx[wave][k] = x[k];
  __syncthreads();
  float yks = proj32(sx[wave], Wks, bks, lane);
  float yvs = proj32(sx[wave], Wvs, bvs, lane);
  float yke = proj32(sx[wave], Wke, bke, lane);
  float yve = proj32(sx[wave], Wve, bve, lane);
  float yko = proj3 (sx[wave], Wko, bko, lane);
  float ks = log_map32(yks);
  float vs = log_map32(yvs);
  float ke = log_map32(yke);
  float ve = log_map32(yve);
  _Float16 keh = (_Float16)ke;
  Ks16[row * 32 + lane] = (_Float16)ks;
  Ke16[row * 32 + lane] = keh;
  VsT16[(b * 32 + lane) * NK_ + key] = (_Float16)vs;   // [B][H][NK]
  VeT16[(b * 32 + lane) * NK_ + key] = (_Float16)ve;
  float kef = (float)keh;
  float s2 = redsum32(kef * kef);
  if (lane == 0) kk[row] = s2;
  float y0 = __shfl(yko, 0, 32), y1 = __shfl(yko, 1, 32), y2 = __shfl(yko, 2, 32);
  float s = y0 * y0 + y1 * y1 + y2 * y2;
  float n = fmaxf(fsqrt(s), 1e-12f);
  float invn = frcp(n);
  if (lane == 0) Ko[row * 4 + 0] = y0 * invn;
  if (lane == 1) Ko[row * 4 + 1] = y1 * invn;
  if (lane == 2) Ko[row * 4 + 2] = y2 * invn;
  if (lane == 3) Ko[row * 4 + 3] = s * invn * invn;
}

// ---------------------------------------------------------------------------
// Kernel 3: pass 1 — sum of electronic distances per (batch, q-block)
// (deterministic: one partial per block, reduced by escale_kernel)
// ---------------------------------------------------------------------------
__global__ void elec_mean_kernel(const _Float16* __restrict__ Qe16,
                                 const _Float16* __restrict__ Ke16,
                                 const float* __restrict__ qq,
                                 const float* __restrict__ kk,
                                 float* __restrict__ partial) {
  __shared__ __align__(16) _Float16 sKe[32 * 32];
  __shared__ float skk[32];
  __shared__ float wpart[4];
  int b = blockIdx.y;
  int q0 = blockIdx.x * 64;
  int wave = threadIdx.x >> 5, lane = threadIdx.x & 31;
  int r = lane & 15, hf = lane >> 4;
  long long qbase = (long long)b * NQ_ + q0 + wave * 16;
  v16h aQe = load_frag16(Qe16 + (qbase + r) * 32, lane);
  float qqf[8];
  #pragma unroll
  for (int i = 0; i < 8; ++i) qqf[i] = qq[qbase + i + 8 * hf];
  float acc = 0.f;
  for (int k0 = 0; k0 < NK_; k0 += 32) {
    __syncthreads();
    {
      int tr = threadIdx.x >> 2;
      int tc = (threadIdx.x & 3) * 8;
      long long kb = (long long)b * NK_ + k0;
      *(v8h*)&sKe[tr * 32 + tc] = *(const v8h*)&Ke16[(kb + tr) * 32 + tc];
      if (threadIdx.x < 32) skk[threadIdx.x] = kk[kb + threadIdx.x];
    }
    __syncthreads();
    #pragma unroll
    for (int t = 0; t < 2; ++t) {
      v16h bKe = load_frag16(&sKe[(r + 16 * t) * 32], lane);
      v8f c = wmma16(aQe, bKe, zero8());
      float kkn = skk[r + 16 * t];
      #pragma unroll
      for (int i = 0; i < 8; ++i) {
        float d2 = qqf[i] + kkn - 2.f * c[i];
        acc += fsqrt(fmaxf(d2, 0.f));
      }
    }
  }
  acc = redsum32(acc);
  if (lane == 0) wpart[wave] = acc;
  __syncthreads();
  if (threadIdx.x == 0)
    partial[b * 32 + blockIdx.x] = wpart[0] + wpart[1] + wpart[2] + wpart[3];
}

__global__ void escale_kernel(const float* __restrict__ partial,
                              float* __restrict__ escale) {
  int b = threadIdx.x;
  if (b < B_) {
    float s = 0.f;
    for (int i = 0; i < 32; ++i) s += partial[b * 32 + i];
    escale[b] = fmaxf(s / (float)((long long)NQ_ * NK_), 1e-4f);
  }
}

// ---------------------------------------------------------------------------
// Kernel 4: fused flash attention (64 q-rows per WG, 32-key tiles)
// Softmax runs in the log2 domain: logit coefficients are pre-scaled by
// log2(e) so every exp/log in the hot loop is a single v_exp_f32/v_log_f32.
// ---------------------------------------------------------------------------
__global__ void attn_kernel(const _Float16* __restrict__ Qs16,
                            const _Float16* __restrict__ Qe16,
                            const float* __restrict__ Qo,
                            const _Float16* __restrict__ Ks16,
                            const _Float16* __restrict__ Ke16,
                            const float* __restrict__ Ko,
                            const _Float16* __restrict__ VsT16,
                            const _Float16* __restrict__ VeT16,
                            const float* __restrict__ qq,
                            const float* __restrict__ kk,
                            const float* __restrict__ escale,
                            const float* __restrict__ pi_star,
                            const float* __restrict__ g_gamma,
                            const float* __restrict__ g_ew,
                            const float* __restrict__ g_ow,
                            float* __restrict__ ctx) {
  __shared__ __align__(16) _Float16 sKs[32 * 32];
  __shared__ __align__(16) _Float16 sKe[32 * 32];
  __shared__ __align__(16) _Float16 sVs[32 * 32];   // transposed-V tile [h][key]
  __shared__ __align__(16) _Float16 sVe[32 * 32];
  __shared__ __align__(16) float sKo[32 * 4];
  __shared__ float skk[32];
  __shared__ __align__(16) _Float16 sP[4][16 * 32]; // per-wave P staging

  int b = blockIdx.y, q0 = blockIdx.x * 64;
  int wave = threadIdx.x >> 5, lane = threadIdx.x & 31;
  int r = lane & 15, hf = lane >> 4;
  // log2-domain logit coefficients
  float gamma  = g_gamma[0];                                   // * log2(pi)
  float c_ss   = 0.17677669529663687f * LOG2E_;                // (1/sqrt(H))*log2e
  float c_el   = -g_ew[0] * frcp(escale[b]) * LOG2E_;          // -e_w/scale*log2e
  float c_or   = g_ow[0] * LOG2E_;                             // o_w*log2e

  long long qbase = (long long)b * NQ_ + q0 + wave * 16;
  v16h aQs = load_frag16(Qs16 + (qbase + r) * 32, lane);
  v16h aQe = load_frag16(Qe16 + (qbase + r) * 32, lane);

  float qqf[8], qof[8][3], qn2f[8], mst[8], lst[8];
  #pragma unroll
  for (int i = 0; i < 8; ++i) {
    long long rr = qbase + i + 8 * hf;
    qqf[i] = qq[rr];
    qof[i][0] = Qo[rr * 4 + 0];
    qof[i][1] = Qo[rr * 4 + 1];
    qof[i][2] = Qo[rr * 4 + 2];
    qn2f[i]   = Qo[rr * 4 + 3];
    mst[i] = -INFINITY;
    lst[i] = 0.f;
  }
  v8f Os0 = zero8(), Os1 = zero8(), Oe0 = zero8(), Oe1 = zero8();

  for (int k0 = 0; k0 < NK_; k0 += 32) {
    __syncthreads();
    {
      int tr = threadIdx.x >> 2;
      int tc = (threadIdx.x & 3) * 8;
      long long kb = (long long)b * NK_ + k0;
      *(v8h*)&sKs[tr * 32 + tc] = *(const v8h*)&Ks16[(kb + tr) * 32 + tc];
      *(v8h*)&sKe[tr * 32 + tc] = *(const v8h*)&Ke16[(kb + tr) * 32 + tc];
      *(v8h*)&sVs[tr * 32 + tc] = *(const v8h*)&VsT16[((long long)b * 32 + tr) * NK_ + k0 + tc];
      *(v8h*)&sVe[tr * 32 + tc] = *(const v8h*)&VeT16[((long long)b * 32 + tr) * NK_ + k0 + tc];
      if (threadIdx.x < 32) {
        skk[threadIdx.x] = kk[kb + threadIdx.x];
        *(float4*)&sKo[threadIdx.x * 4] = *(const float4*)&Ko[(kb + threadIdx.x) * 4];
      }
    }
    __syncthreads();

    v8f L0, L1;
    #pragma unroll
    for (int t = 0; t < 2; ++t) {
      v16h bKs = load_frag16(&sKs[(r + 16 * t) * 32], lane);
      v16h bKe = load_frag16(&sKe[(r + 16 * t) * 32], lane);
      v8f Ss = wmma16(aQs, bKs, zero8());
      v8f Se = wmma16(aQe, bKe, zero8());
      int n = r + 16 * t;
      float kkn = skk[n];
      float ko0 = sKo[n * 4 + 0], ko1 = sKo[n * 4 + 1];
      float ko2 = sKo[n * 4 + 2], kn2 = sKo[n * 4 + 3];
      const float* prow =
          pi_star + ((long long)b * NQ_ + q0 + wave * 16 + 8 * hf) * NK_ + k0 + n;
      if (k0 + 32 < NK_) __builtin_prefetch(prow + 32, 0, 1);  // next pi tile
      v8f Lt;
      #pragma unroll
      for (int i = 0; i < 8; ++i) {
        float d2 = qqf[i] + kkn - 2.f * Se[i];
        float el = fsqrt(fmaxf(d2, 0.f));                     // distance
        float dot = qof[i][0] * ko0 + qof[i][1] * ko1 + qof[i][2] * ko2;
        float wp  = fsqrt(fmaxf(qn2f[i] * kn2 - dot * dot, 0.f));
        float pv  = prow[(long long)i * NK_];
        float pl  = gamma * flog2(fmaxf(pv, 1e-9f));          // log2-domain mask
        Lt[i] = Ss[i] * c_ss + el * c_el + (dot - wp) * c_or + pl;
      }
      if (t == 0) L0 = Lt; else L1 = Lt;
    }

    // online softmax update in log2 domain (rows i + 8*hf per lane-half)
    #pragma unroll
    for (int i = 0; i < 8; ++i) {
      float rm = redmax16(fmaxf(L0[i], L1[i]));
      float mn = fmaxf(mst[i], rm);
      float a  = fexp2(mst[i] - mn);
      float p0 = fexp2(L0[i] - mn);
      float p1 = fexp2(L1[i] - mn);
      lst[i] = lst[i] * a + redsum16(p0 + p1);
      mst[i] = mn;
      sP[wave][(i + 8 * hf) * 32 + r]      = (_Float16)p0;
      sP[wave][(i + 8 * hf) * 32 + 16 + r] = (_Float16)p1;
      Os0[i] *= a; Os1[i] *= a; Oe0[i] *= a; Oe1[i] *= a;
    }
    __syncthreads();  // order sP writes vs fragment reads

    v16h aP   = load_frag16(&sP[wave][r * 32], lane);
    v16h bVs0 = load_frag16(&sVs[r * 32], lane);
    v16h bVs1 = load_frag16(&sVs[(r + 16) * 32], lane);
    v16h bVe0 = load_frag16(&sVe[r * 32], lane);
    v16h bVe1 = load_frag16(&sVe[(r + 16) * 32], lane);
    Os0 = wmma16(aP, bVs0, Os0);
    Os1 = wmma16(aP, bVs1, Os1);
    Oe0 = wmma16(aP, bVe0, Oe0);
    Oe1 = wmma16(aP, bVe1, Oe1);
  }

  #pragma unroll
  for (int i = 0; i < 8; ++i) {
    float invl = frcp(lst[i]);
    float* dst = ctx + ((long long)b * NQ_ + q0 + wave * 16 + i + 8 * hf) * 64;
    dst[r]      = Os0[i] * invl;
    dst[16 + r] = Os1[i] * invl;
    dst[32 + r] = Oe0[i] * invl;
    dst[48 + r] = Oe1[i] * invl;
  }
}

// ---------------------------------------------------------------------------
// Kernel 5: fused epilogue MLP (concat -> W1 -> SiLU -> W2)
// ---------------------------------------------------------------------------
__global__ void mlp_kernel(const float* __restrict__ ctx,
                           const float* __restrict__ W1, const float* __restrict__ b1,
                           const float* __restrict__ W2, const float* __restrict__ b2,
                           float* __restrict__ out) {
  __shared__ float sfu[4][128];
  __shared__ float sh1[4][32];
  int wave = threadIdx.x >> 5, lane = threadIdx.x & 31;
  long long row = (long long)blockIdx.x * 4 + wave;
  const float* c = ctx + row * 64;
  float cs = c[lane], ce = c[32 + lane];
  sfu[wave][lane]      = cs;
  sfu[wave][32 + lane] = ce;
  sfu[wave][64 + lane] = cs - ce;
  sfu[wave][96 + lane] = cs * ce;
  __syncthreads();
  float h = b1[lane];
  #pragma unroll 8
  for (int k = 0; k < 128; ++k)
    h = fmaf(sfu[wave][k], W1[k * 32 + lane], h);
  h = h * frcp(1.0f + fexp2(-LOG2E_ * h));  // SiLU
  sh1[wave][lane] = h;
  __syncthreads();
  float o = b2[lane];
  #pragma unroll 8
  for (int k = 0; k < 32; ++k)
    o = fmaf(sh1[wave][k], W2[k * 32 + lane], o);
  out[row * 32 + lane] = o;
}

// ---------------------------------------------------------------------------
extern "C" void kernel_launch(void* const* d_in, const int* in_sizes, int n_in,
                              void* d_out, int out_size, void* d_ws, size_t ws_size,
                              hipStream_t stream) {
  const float* q_fp    = (const float*)d_in[0];
  const float* v_ret   = (const float*)d_in[1];
  const float* pi_star = (const float*)d_in[2];
  const float* Wqs = (const float*)d_in[3];  const float* bqs = (const float*)d_in[4];
  const float* Wks = (const float*)d_in[5];  const float* bks = (const float*)d_in[6];
  const float* Wvs = (const float*)d_in[7];  const float* bvs = (const float*)d_in[8];
  const float* Wqe = (const float*)d_in[9];  const float* bqe = (const float*)d_in[10];
  const float* Wke = (const float*)d_in[11]; const float* bke = (const float*)d_in[12];
  const float* Wve = (const float*)d_in[13]; const float* bve = (const float*)d_in[14];
  const float* Wqo = (const float*)d_in[15]; const float* bqo = (const float*)d_in[16];
  const float* Wko = (const float*)d_in[17]; const float* bko = (const float*)d_in[18];
  const float* W1  = (const float*)d_in[19]; const float* b1  = (const float*)d_in[20];
  const float* W2  = (const float*)d_in[21]; const float* b2  = (const float*)d_in[22];
  const float* gamma = (const float*)d_in[23];
  const float* e_w   = (const float*)d_in[24];
  const float* o_w   = (const float*)d_in[25];

  char* ws = (char*)d_ws;
  size_t off = 0;
  auto alloc = [&](size_t bytes) -> void* {
    void* p = ws + off;
    off += (bytes + 255) & ~(size_t)255;
    return p;
  };
  _Float16* Qs16  = (_Float16*)alloc((size_t)B_ * NQ_ * 32 * sizeof(_Float16));
  _Float16* Qe16  = (_Float16*)alloc((size_t)B_ * NQ_ * 32 * sizeof(_Float16));
  _Float16* Ks16  = (_Float16*)alloc((size_t)B_ * NK_ * 32 * sizeof(_Float16));
  _Float16* Ke16  = (_Float16*)alloc((size_t)B_ * NK_ * 32 * sizeof(_Float16));
  _Float16* VsT16 = (_Float16*)alloc((size_t)B_ * 32 * NK_ * sizeof(_Float16));
  _Float16* VeT16 = (_Float16*)alloc((size_t)B_ * 32 * NK_ * sizeof(_Float16));
  float* Qo      = (float*)alloc((size_t)B_ * NQ_ * 4 * sizeof(float));
  float* Ko      = (float*)alloc((size_t)B_ * NK_ * 4 * sizeof(float));
  float* qq      = (float*)alloc((size_t)B_ * NQ_ * sizeof(float));
  float* kk      = (float*)alloc((size_t)B_ * NK_ * sizeof(float));
  float* partial = (float*)alloc((size_t)B_ * 32 * sizeof(float));
  float* escale  = (float*)alloc((size_t)B_ * sizeof(float));
  float* ctx     = (float*)alloc((size_t)B_ * NQ_ * 64 * sizeof(float));

  proj_q_kernel<<<dim3(B_ * NQ_ / 4), dim3(128), 0, stream>>>(
      q_fp, Wqs, bqs, Wqe, bqe, Wqo, bqo, Qs16, Qe16, Qo, qq);
  proj_kv_kernel<<<dim3(B_ * NK_ / 4), dim3(128), 0, stream>>>(
      v_ret, Wks, bks, Wvs, bvs, Wke, bke, Wve, bve, Wko, bko,
      Ks16, Ke16, VsT16, VeT16, Ko, kk);
  elec_mean_kernel<<<dim3(NQ_ / 64, B_), dim3(128), 0, stream>>>(
      Qe16, Ke16, qq, kk, partial);
  escale_kernel<<<dim3(1), dim3(32), 0, stream>>>(partial, escale);
  attn_kernel<<<dim3(NQ_ / 64, B_), dim3(128), 0, stream>>>(
      Qs16, Qe16, Qo, Ks16, Ke16, Ko, VsT16, VeT16, qq, kk, escale,
      pi_star, gamma, e_w, o_w, ctx);
  mlp_kernel<<<dim3(B_ * NQ_ / 4), dim3(128), 0, stream>>>(
      ctx, W1, b1, W2, b2, (float*)d_out);
}
